// MultiContextAttention_21861383536769
// MI455X (gfx1250) — compile-verified, matrix operand order checked by reading, not run on hardware
//
#include <hip/hip_runtime.h>

#define EMB   1024
#define HEADS 16
#define HDIM  64
#define NBAT  4
#define TIQ   512
#define TTOT  1024

typedef __bf16 bf16_t;
typedef bf16_t v16bf __attribute__((ext_vector_type(16)));
typedef bf16_t v8bf  __attribute__((ext_vector_type(8)));
typedef bf16_t v4bf  __attribute__((ext_vector_type(4)));
typedef float  v8f   __attribute__((ext_vector_type(8)));

// ---------------------------------------------------------------------------
// Tensor Data Mover (6-arg clang-23 form, manual staging fallback)
// ---------------------------------------------------------------------------
#if __has_builtin(__builtin_amdgcn_tensor_load_to_lds) && __has_builtin(__builtin_amdgcn_s_wait_tensorcnt)
#define USE_TDM 1
typedef unsigned int u32x4 __attribute__((ext_vector_type(4)));
typedef int          i32x8 __attribute__((ext_vector_type(8)));
typedef int          i32x4 __attribute__((ext_vector_type(4)));

// 2D bf16 tile load: tile_rows x tile_cols elements, row stride = stride0 elems.
// D# bitfields per CDNA5 ISA §8.3/8.4 (count=1, type=2, data_size=1 -> 2 bytes).
__device__ __forceinline__ void tdm_load_2d(unsigned int lds_off, const void* gptr,
                                            unsigned int tile_rows, unsigned int tile_cols,
                                            unsigned int tdim0, unsigned int tdim1,
                                            unsigned int stride0)
{
    unsigned long long ga = (unsigned long long)(uintptr_t)gptr;
    u32x4 g0 = { 1u,                                   // count=1, user mode
                 lds_off,                              // lds_addr
                 (unsigned int)ga,                     // global_addr[31:0]
                 (unsigned int)((ga >> 32) & 0x01FFFFFFu) | (2u << 30) }; // [56:32] | type=2
    i32x8 g1;
    g1[0] = (int)(1u << 16);                           // wg_mask=0, data_size=1 (2B)
    g1[1] = (int)(tdim0 << 16);                        // tensor_dim0[15:0]
    g1[2] = (int)(((tdim0 >> 16) & 0xFFFFu) | (tdim1 << 16));      // dim0 hi | dim1 lo
    g1[3] = (int)(((tdim1 >> 16) & 0xFFFFu) | (tile_cols << 16));  // dim1 hi | tile_dim0
    g1[4] = (int)tile_rows;                            // tile_dim1 (tile_dim2=0)
    g1[5] = (int)stride0;                              // tensor_dim0_stride[31:0]
    g1[6] = 0;
    g1[7] = 0;
    i32x4 z4 = {0, 0, 0, 0};                           // groups 2/3 unused (2D)
    i32x8 z8 = {0, 0, 0, 0, 0, 0, 0, 0};
    __builtin_amdgcn_tensor_load_to_lds(g0, g1, z4, z4, z8, 0);
}
#endif

// ISA 7.12.2: A/B 16-bit fragment of lane l (half = l>>4) is two contiguous
// 8-element k-runs: [half*8, half*8+8) and [16+half*8, ...). Load as 2x b128.
__device__ __forceinline__ v16bf ld_frag(const bf16_t* base, int half) {
    v8bf lo = *(const v8bf*)(base + (half << 3));
    v8bf hi = *(const v8bf*)(base + 16 + (half << 3));
    return __builtin_shufflevector(lo, hi, 0, 1, 2, 3, 4, 5, 6, 7,
                                           8, 9, 10, 11, 12, 13, 14, 15);
}

__device__ __forceinline__ v8f wmma_bf16(v16bf a, v16bf b, v8f c) {
    return __builtin_amdgcn_wmma_f32_16x16x32_bf16(false, a, false, b,
                                                   (short)0, c, false, false);
}

// ---------------------------------------------------------------------------
// fp32 -> bf16 bulk convert (n4 = n/4 float4 groups)
// ---------------------------------------------------------------------------
__global__ __launch_bounds__(256) void cvt_bf16_kernel(const float* __restrict__ src,
                                                       bf16_t* __restrict__ dst, int n4)
{
    int i = blockIdx.x * 256 + threadIdx.x;
    if (i < n4) {
        float4 v = ((const float4*)src)[i];
        v4bf o;
        o[0] = (bf16_t)v.x; o[1] = (bf16_t)v.y;
        o[2] = (bf16_t)v.z; o[3] = (bf16_t)v.w;
        ((v4bf*)dst)[i] = o;
    }
}

// ---------------------------------------------------------------------------
// GEMM: C[m][n] = sum_k A[m][k]*W[n][k] (+bias). A,W bf16; out bf16 or fp32.
// 8 waves/block -> 128x64 tile; W tile (64n x 32k) staged in LDS via TDM.
// ---------------------------------------------------------------------------
__global__ __launch_bounds__(256) void gemm_wmma_kernel(
    const bf16_t* __restrict__ A, const bf16_t* __restrict__ W,
    const float* __restrict__ bias, void* __restrict__ out,
    int out_bf16, int Tin, int t_off, long long out_bstride)
{
    __shared__ bf16_t sW[64 * 32];             // [n][k], k contiguous

    const int tid  = threadIdx.x;
    const int w    = tid >> 5;
    const int lane = tid & 31;
    const int lid  = lane & 15;
    const int half = lane >> 4;

    const int m0 = blockIdx.x * 128 + w * 16;
    const int n0 = blockIdx.y * 64;

    v8f acc[4] = {};
    const bf16_t* arow = A + (size_t)(m0 + lid) * EMB;

    for (int k0 = 0; k0 < EMB; k0 += 32) {
        __syncthreads();                       // previous tile fully consumed
#ifdef USE_TDM
        if (tid < 32) {
            tdm_load_2d((unsigned int)(uintptr_t)&sW[0], W + (size_t)n0 * EMB + k0,
                        /*rows*/64, /*cols*/32, /*tdim0*/32, /*tdim1*/64, /*stride*/EMB);
            __builtin_amdgcn_s_wait_tensorcnt(0);
        }
#else
        {   // fallback: one v8bf copy per thread
            const int n  = tid >> 2;           // 0..63
            const int kk = (tid & 3) * 8;      // 0,8,16,24
            const bf16_t* wp = W + (size_t)(n0 + n) * EMB + k0 + kk;
            if (k0 + 32 < EMB) __builtin_prefetch(wp + 32, 0, 0);
            *(v8bf*)(sW + n * 32 + kk) = *(const v8bf*)wp;
        }
#endif
        __syncthreads();

        v16bf afrag = ld_frag(arow + k0, half);        // 2x global b128
        #pragma unroll
        for (int nb = 0; nb < 4; ++nb) {
            v16bf bfrag = ld_frag(sW + (nb * 16 + lid) * 32, half); // 2x ds b128
            acc[nb] = wmma_bf16(afrag, bfrag, acc[nb]);
        }
    }

    #pragma unroll
    for (int nb = 0; nb < 4; ++nb) {
        #pragma unroll
        for (int r = 0; r < 8; ++r) {
            const int m  = m0 + r + half * 8;
            const int bi = m / Tin;
            const int t  = m - bi * Tin;
            const int n  = n0 + nb * 16 + lid;
            float v = acc[nb][r];
            if (bias) v += bias[n];
            const size_t off = (size_t)bi * out_bstride + (size_t)(t + t_off) * EMB + n;
            if (out_bf16) ((bf16_t*)out)[off] = (bf16_t)v;
            else          ((float*)out)[off]  = v;
        }
    }
}

// ---------------------------------------------------------------------------
// Flash attention: 4 waves/block, each wave one 16-query tile of one (b,h).
// K chunk (32 keys x 64 d) staged row-major via TDM; V staged TRANSPOSED
// ([d][key]) so P·V B-fragments are contiguous b128 loads.
// ---------------------------------------------------------------------------
__global__ __launch_bounds__(128) void attn_wmma_kernel(
    const bf16_t* __restrict__ Q, const bf16_t* __restrict__ K,
    const bf16_t* __restrict__ V, bf16_t* __restrict__ AO)
{
    __shared__ bf16_t sK[32 * 64];             // [key][d]
    __shared__ bf16_t sVt[64 * 32];            // [d][key]  (transposed)
    __shared__ bf16_t sP[4][16 * 32];          // per-wave P tile, [q][key]

    const int tid  = threadIdx.x;
    const int w    = tid >> 5;
    const int lane = tid & 31;
    const int lid  = lane & 15;
    const int half = lane >> 4;

    const int h  = blockIdx.y;
    const int b  = blockIdx.z;
    const int q0 = (blockIdx.x * 4 + w) * 16;

    const bf16_t* qrow = Q + ((size_t)(b * TIQ + q0 + lid)) * EMB + h * HDIM;
    v16bf aq0 = ld_frag(qrow, half);
    v16bf aq1 = ld_frag(qrow + 32, half);

    float mrow[8], lrow[8];
    #pragma unroll
    for (int r = 0; r < 8; ++r) { mrow[r] = -1e30f; lrow[r] = 0.f; }
    v8f oacc[4] = {};

    const float scale = 0.03125f;              // 1/sqrt(EMB)

    const int crow = tid >> 2;                 // 0..31 key row
    const int ccol = (tid & 3) * 16;           // 0,16,32,48 d offset

    for (int kt = 0; kt < TTOT / 32; ++kt) {
        __syncthreads();                       // prior iteration done with LDS
        const size_t gRow = ((size_t)(b * TTOT + kt * 32)) * EMB + h * HDIM;

#ifdef USE_TDM
        if (tid < 32) {
            tdm_load_2d((unsigned int)(uintptr_t)&sK[0], K + gRow,
                        /*rows*/32, /*cols*/64, /*tdim0*/64, /*tdim1*/32, /*stride*/EMB);
        }
#else
        {   // fallback K staging: 32B per thread
            const bf16_t* kp = K + gRow + (size_t)crow * EMB + ccol;
            *(v8bf*)(sK + crow * 64 + ccol)     = *(const v8bf*)kp;
            *(v8bf*)(sK + crow * 64 + ccol + 8) = *(const v8bf*)(kp + 8);
            if (kt + 1 < TTOT / 32) __builtin_prefetch(kp + 32 * EMB, 0, 0);
        }
#endif
        {   // V staged transposed: coalesced global read, scattered LDS write
            const bf16_t* vp = V + gRow + (size_t)crow * EMB + ccol;
            v8bf a0 = *(const v8bf*)vp;
            v8bf a1 = *(const v8bf*)(vp + 8);
            #pragma unroll
            for (int i = 0; i < 8; ++i) sVt[(ccol + i) * 32 + crow]     = a0[i];
            #pragma unroll
            for (int i = 0; i < 8; ++i) sVt[(ccol + 8 + i) * 32 + crow] = a1[i];
            if (kt + 1 < TTOT / 32) __builtin_prefetch(vp + 32 * EMB, 0, 0);
        }
#ifdef USE_TDM
        if (tid < 32) __builtin_amdgcn_s_wait_tensorcnt(0);
#endif
        __syncthreads();

        // S = Q K^T : two 16x16 key tiles, 2 WMMAs each over d
        v8f s0 = {}, s1 = {};
        s0 = wmma_bf16(aq0, ld_frag(sK + lid * 64,             half), s0);
        s0 = wmma_bf16(aq1, ld_frag(sK + lid * 64 + 32,        half), s0);
        s1 = wmma_bf16(aq0, ld_frag(sK + (16 + lid) * 64,      half), s1);
        s1 = wmma_bf16(aq1, ld_frag(sK + (16 + lid) * 64 + 32, half), s1);

        // online softmax over 32 new keys (masks all-ones -> no floor branch)
        #pragma unroll
        for (int r = 0; r < 8; ++r) {
            float x0 = s0[r] * scale, x1 = s1[r] * scale;
            float mx = fmaxf(x0, x1);
            #pragma unroll
            for (int off = 8; off >= 1; off >>= 1)
                mx = fmaxf(mx, __shfl_xor(mx, off, 32));
            float mn    = fmaxf(mrow[r], mx);
            float alpha = __expf(mrow[r] - mn);
            float p0 = __expf(x0 - mn), p1 = __expf(x1 - mn);
            float ps = p0 + p1;
            #pragma unroll
            for (int off = 8; off >= 1; off >>= 1)
                ps += __shfl_xor(ps, off, 32);
            lrow[r] = lrow[r] * alpha + ps;
            mrow[r] = mn;
            #pragma unroll
            for (int nb = 0; nb < 4; ++nb) oacc[nb][r] *= alpha;
            sP[w][(r + half * 8) * 32 + lid]      = (bf16_t)p0;
            sP[w][(r + half * 8) * 32 + 16 + lid] = (bf16_t)p1;
        }

        // P as A fragment (row-major in LDS -> 2x ds b128)
        v16bf aP = ld_frag(sP[w] + lid * 32, half);

        // O += P V : four 16-wide d tiles from transposed V
        #pragma unroll
        for (int nb = 0; nb < 4; ++nb) {
            v16bf bv = ld_frag(sVt + (nb * 16 + lid) * 32, half);
            oacc[nb] = wmma_bf16(aP, bv, oacc[nb]);
        }
    }

    #pragma unroll
    for (int nb = 0; nb < 4; ++nb) {
        #pragma unroll
        for (int r = 0; r < 8; ++r) {
            const size_t off = ((size_t)(b * TIQ + q0 + r + half * 8)) * EMB
                               + h * HDIM + nb * 16 + lid;
            AO[off] = (bf16_t)(oacc[nb][r] / lrow[r]);
        }
    }
}

// ---------------------------------------------------------------------------
extern "C" void kernel_launch(void* const* d_in, const int* in_sizes, int n_in,
                              void* d_out, int out_size, void* d_ws, size_t ws_size,
                              hipStream_t stream)
{
    (void)in_sizes; (void)n_in; (void)out_size; (void)ws_size;

    const float* x    = (const float*)d_in[0];
    const float* c0   = (const float*)d_in[1];
    const float* c1   = (const float*)d_in[2];
    const float* Wsrc[8] = {
        (const float*)d_in[6],  (const float*)d_in[7],  (const float*)d_in[8],
        (const float*)d_in[9],  (const float*)d_in[10], (const float*)d_in[11],
        (const float*)d_in[12], (const float*)d_in[13] };  // q,k,v,ck0,ck1,cv0,cv1,u
    const float* bu = (const float*)d_in[14];

    const size_t nx = (size_t)NBAT * TIQ * EMB;      // 2M elems
    const size_t nc = (size_t)NBAT * 256 * EMB;      // 1M
    const size_t nw = (size_t)EMB * EMB;             // 1M
    const size_t nk = (size_t)NBAT * TTOT * EMB;     // 4M

    bf16_t* xb  = (bf16_t*)d_ws;
    bf16_t* c0b = xb  + nx;
    bf16_t* c1b = c0b + nc;
    bf16_t* Wb  = c1b + nc;                          // 8 weight matrices
    bf16_t* Qw  = Wb  + 8 * nw;
    bf16_t* Kw  = Qw  + nx;
    bf16_t* Vw  = Kw  + nk;
    bf16_t* AOb = Vw  + nk;                          // total 48 MB

    auto cvt = [&](const float* s, bf16_t* d, size_t n) {
        int n4 = (int)(n / 4);
        cvt_bf16_kernel<<<dim3((n4 + 255) / 256), dim3(256), 0, stream>>>(s, d, n4);
    };
    cvt(x, xb, nx);
    cvt(c0, c0b, nc);
    cvt(c1, c1b, nc);
    for (int i = 0; i < 8; ++i) cvt(Wsrc[i], Wb + (size_t)i * nw, nw);

    const dim3 blk(256);
    const long long bsQ = (long long)TIQ  * EMB;
    const long long bsK = (long long)TTOT * EMB;

    gemm_wmma_kernel<<<dim3(16,16), blk, 0, stream>>>(xb,  Wb + 0*nw, nullptr, Qw, 1, TIQ, 0,   bsQ);
    gemm_wmma_kernel<<<dim3(16,16), blk, 0, stream>>>(xb,  Wb + 1*nw, nullptr, Kw, 1, TIQ, 0,   bsK);
    gemm_wmma_kernel<<<dim3(8,16),  blk, 0, stream>>>(c0b, Wb + 3*nw, nullptr, Kw, 1, 256, 512, bsK);
    gemm_wmma_kernel<<<dim3(8,16),  blk, 0, stream>>>(c1b, Wb + 4*nw, nullptr, Kw, 1, 256, 768, bsK);
    gemm_wmma_kernel<<<dim3(16,16), blk, 0, stream>>>(xb,  Wb + 2*nw, nullptr, Vw, 1, TIQ, 0,   bsK);
    gemm_wmma_kernel<<<dim3(8,16),  blk, 0, stream>>>(c0b, Wb + 5*nw, nullptr, Vw, 1, 256, 512, bsK);
    gemm_wmma_kernel<<<dim3(8,16),  blk, 0, stream>>>(c1b, Wb + 6*nw, nullptr, Vw, 1, 256, 768, bsK);

    attn_wmma_kernel<<<dim3(TIQ/64, HEADS, NBAT), dim3(128), 0, stream>>>(Qw, Kw, Vw, AOb);

    gemm_wmma_kernel<<<dim3(16,16), blk, 0, stream>>>(AOb, Wb + 7*nw, bu, d_out, 0, TIQ, 0, bsQ);
}
